// Attention_24816321036632
// MI455X (gfx1250) — matrix operand order repaired and checked by reading, not compiled, not run on hardware
//
#include <hip/hip_runtime.h>

#define DIM    1024
#define HEADS  16
#define DHEAD  64
#define NTOK   2048
#define BATCH  2
#define MROWS  (BATCH * NTOK)      // 4096 tokens
#define SCALEF 0.125f              // 64^-0.5

typedef __attribute__((ext_vector_type(16))) _Float16 v16h;
typedef __attribute__((ext_vector_type(8)))  float    v8f;

union Frag { uint4 u[2]; v16h v; };

#if defined(__has_builtin)
#if __has_builtin(__builtin_amdgcn_global_load_async_to_lds_b128)
#define HAVE_ASYNC_LDS 1
#endif
#endif

typedef int v4i32 __attribute__((__vector_size__(16)));
typedef __attribute__((address_space(1))) v4i32* gv4p;   // global int4*
typedef __attribute__((address_space(3))) v4i32* lv4p;   // LDS int4*

// Stage 16 bytes global -> LDS. Async (ASYNCcnt-tracked) when available.
static __device__ inline void stage16(const _Float16* g, _Float16* l) {
#ifdef HAVE_ASYNC_LDS
  __builtin_amdgcn_global_load_async_to_lds_b128((gv4p)g, (lv4p)l, 0, 0);
#else
  *reinterpret_cast<uint4*>(l) = *reinterpret_cast<const uint4*>(g);
#endif
}

static __device__ inline void async_wait() {
#ifdef HAVE_ASYNC_LDS
#if __has_builtin(__builtin_amdgcn_s_wait_asynccnt)
  __builtin_amdgcn_s_wait_asynccnt(0);
#else
  asm volatile("s_wait_asynccnt 0x0" ::: "memory");
#endif
#endif
}

// A-fragment (16x32 f16): lane = row (lane&15); VGPR j holds
// K = 16*(j>>2) + 8*(lane>>4) + 2*(j&3)  -> two contiguous 16B loads
// at element offsets 8h and 16+8h from a row-major [M][Ktot] matrix.
static __device__ inline v16h load_frag_a(const _Float16* p, int stride,
                                          int row, int kbase, int lane) {
  int r = lane & 15, h = (lane >> 4) & 1;
  const _Float16* base = p + (size_t)(row + r) * stride + kbase + 8 * h;
  Frag f;
  f.u[0] = *reinterpret_cast<const uint4*>(base);
  f.u[1] = *reinterpret_cast<const uint4*>(base + 16);
  return f.v;
}

// B-fragment (32x16 f16): lane = col n (lane&15); VGPR j holds
// K = 16*(j>>2) + 8*(lane>>4) + 2*(j&3) -> two contiguous 16B loads at element offsets
// 8h and 16+8h from a matrix stored [n][k] (k contiguous, stride elems).
static __device__ inline v16h load_frag_b(const _Float16* p, int stride,
                                          int nrow, int kbase, int lane) {
  int r = lane & 15, h = (lane >> 4) & 1;
  const _Float16* base = p + (size_t)(nrow + r) * stride + kbase + 8 * h;
  Frag f;
  f.u[0] = *reinterpret_cast<const uint4*>(base);
  f.u[1] = *reinterpret_cast<const uint4*>(base + 16);
  return f.v;
}

static __device__ inline v8f wmma16(v16h a, v16h b, v8f c) {
  return __builtin_amdgcn_wmma_f32_16x16x32_f16(false, a, false, b,
                                                (short)0, c, false, false);
}

// ---------------- RMSNorm + f32->f16 ----------------
__global__ void __launch_bounds__(256)
rmsnorm_kernel(const float* __restrict__ x, const float* __restrict__ gamma,
               _Float16* __restrict__ xn) {
  int row = blockIdx.x;                       // 4096 rows
  const float* xr = x + (size_t)row * DIM;
  float s = 0.f;
  for (int i = threadIdx.x; i < DIM; i += 256) { float v = xr[i]; s += v * v; }
  s += __shfl_xor(s, 16, 32); s += __shfl_xor(s, 8, 32);
  s += __shfl_xor(s, 4, 32);  s += __shfl_xor(s, 2, 32);
  s += __shfl_xor(s, 1, 32);
  __shared__ float red[8];
  if ((threadIdx.x & 31) == 0) red[threadIdx.x >> 5] = s;
  __syncthreads();
  float tot = red[0] + red[1] + red[2] + red[3] +
              red[4] + red[5] + red[6] + red[7];
  float inv = 32.0f / fmaxf(sqrtf(tot), 1e-12f);   // sqrt(DIM)=32
  for (int i = threadIdx.x; i < DIM; i += 256)
    xn[(size_t)row * DIM + i] = (_Float16)(xr[i] * inv * gamma[i]);
}

// ---------------- weight transpose+convert: WT[n][k] = W[k][n] --------------
__global__ void __launch_bounds__(256)
transpose_kernel(const float* __restrict__ W, _Float16* __restrict__ WT,
                 int K, int N) {
  int idx = blockIdx.x * 256 + threadIdx.x;
  if (idx >= K * N) return;
  int n = idx / K, kk = idx - n * K;
  WT[idx] = (_Float16)W[(size_t)kk * N + n];
}

// ---------------- block-cooperative WMMA GEMM --------------------------------
// C[M][Ntot] = A[M][Ktot] * BT[Ntot][Ktot]^T
// Block = 8 waves = 256x64 output tile; each wave computes 32x64.
// B k-slab (64 cols x 32 k, 4KB f16) staged per block into LDS via the async
// global->LDS path, double buffered. K-loop unrolled by 2 with explicit
// ping-pong of LDS buffers and A fragments (no register-rotation moves).
// mode 0: f16 -> Q[b][h][n][d]
// mode 1: f16 -> K[b][h][n][d] (col<1024) / Vt[b][h][d][n] (col>=1024)
// mode 2: f32 -> outF row-major
__global__ void __launch_bounds__(256)
gemm_kernel(const _Float16* __restrict__ A, const _Float16* __restrict__ BT,
            float* __restrict__ outF, _Float16* __restrict__ out0,
            _Float16* __restrict__ out1, int Ktot, int Ntot, int mode) {
  __shared__ _Float16 bsh[2][64 * 32];     // double-buffered B tile, 2 x 4KB
  int tid = threadIdx.x;
  int lane = tid & 31, wid = tid >> 5;
  int nChunks = Ntot >> 6;
  int mB = blockIdx.x / nChunks, nB = blockIdx.x % nChunks;
  int nBase = nB * 64;
  int mBase = mB * 256 + wid * 32;

  // staging map: thread -> (row 0..63 of B tile, 16B segment 0..3)
  int srow = tid >> 2, sseg = tid & 3;
  const _Float16* bsrc = BT + (size_t)(nBase + srow) * Ktot + sseg * 8;
  _Float16* bdst0 = &bsh[0][srow * 32 + sseg * 8];
  _Float16* bdst1 = &bsh[1][srow * 32 + sseg * 8];

  stage16(bsrc, bdst0);                    // kb = 0 -> buffer 0
  async_wait();
  __syncthreads();

  v8f acc[2][4] = {};
  v16h va0 = load_frag_a(A, Ktot, mBase,      0, lane);
  v16h va1 = load_frag_a(A, Ktot, mBase + 16, 0, lane);
  v16h wa0, wa1;

  for (int kb = 0; kb < Ktot; kb += 64) {   // Ktot % 64 == 0
    // ---- step 0: consume bsh[0] + va*, stage kb+32 -> bsh[1], load wa*
    stage16(bsrc + kb + 32, bdst1);
    wa0 = load_frag_a(A, Ktot, mBase,      kb + 32, lane);
    wa1 = load_frag_a(A, Ktot, mBase + 16, kb + 32, lane);
#pragma unroll
    for (int t = 0; t < 4; ++t) {
      v16h vb = load_frag_b(&bsh[0][0], 32, t * 16, 0, lane);
      acc[0][t] = wmma16(va0, vb, acc[0][t]);
      acc[1][t] = wmma16(va1, vb, acc[1][t]);
    }
    async_wait();
    __syncthreads();

    // ---- step 1: consume bsh[1] + wa*, stage kb+64 -> bsh[0], load va*
    bool more = (kb + 64) < Ktot;
    if (more) {
      stage16(bsrc + kb + 64, bdst0);
      va0 = load_frag_a(A, Ktot, mBase,      kb + 64, lane);
      va1 = load_frag_a(A, Ktot, mBase + 16, kb + 64, lane);
    }
#pragma unroll
    for (int t = 0; t < 4; ++t) {
      v16h vb = load_frag_b(&bsh[1][0], 32, t * 16, 0, lane);
      acc[0][t] = wmma16(wa0, vb, acc[0][t]);
      acc[1][t] = wmma16(wa1, vb, acc[1][t]);
    }
    async_wait();
    __syncthreads();
  }

  int half = lane >> 4, nloc = lane & 15;
#pragma unroll
  for (int sub = 0; sub < 2; ++sub) {
#pragma unroll
    for (int t = 0; t < 4; ++t) {
#pragma unroll
      for (int r = 0; r < 8; ++r) {
        int row = mBase + sub * 16 + r + 8 * half;
        int col = nBase + t * 16 + nloc;
        float v = acc[sub][t][r];
        if (mode == 2) {
          outF[(size_t)row * Ntot + col] = v;
        } else if (mode == 0) {
          int bb2 = row >> 11, n = row & (NTOK - 1);
          int hh = col >> 6, d = col & 63;
          out0[(((size_t)(bb2 * HEADS + hh)) * NTOK + n) * DHEAD + d] = (_Float16)v;
        } else {
          int bb2 = row >> 11, n = row & (NTOK - 1);
          if (col < DIM) {
            int hh = col >> 6, d = col & 63;
            out0[(((size_t)(bb2 * HEADS + hh)) * NTOK + n) * DHEAD + d] = (_Float16)v;
          } else {
            int c = col - DIM;
            int hh = c >> 6, d = c & 63;
            out1[(((size_t)(bb2 * HEADS + hh)) * DHEAD + d) * NTOK + n] = (_Float16)v;
          }
        }
      }
    }
  }
}

// ---------------- flash attention: one wave = 16-query tile of one (b,h) ----
__global__ void __launch_bounds__(256)
attn_kernel(const _Float16* __restrict__ Q, const _Float16* __restrict__ Kb,
            const _Float16* __restrict__ Vt, const unsigned char* __restrict__ mask,
            _Float16* __restrict__ O) {
  __shared__ _Float16 psh[8 * 16 * 32];        // 1KB P-staging per wave
  int lane = threadIdx.x & 31;
  int wid  = threadIdx.x >> 5;
  int gw   = blockIdx.x * 8 + wid;             // 4096 waves
  int mT   = gw & 127;                         // 128 query tiles per (b,h)
  int bh   = gw >> 7;                          // 0..31
  int b = bh >> 4, h = bh & 15;
  int mBase = mT * 16;
  int half = lane >> 4, nloc = lane & 15;

  const _Float16* q  = Q  + (size_t)bh * NTOK * DHEAD;
  const _Float16* k  = Kb + (size_t)bh * NTOK * DHEAD;
  const _Float16* vt = Vt + (size_t)bh * DHEAD * NTOK;
  const unsigned char* mk = mask + b * NTOK;
  _Float16* pw = &psh[wid * 512];

  v16h qa0 = load_frag_a(q, DHEAD, mBase, 0, lane);
  v16h qa1 = load_frag_a(q, DHEAD, mBase, 32, lane);

  v8f acc0 = {}, acc1 = {}, acc2 = {}, acc3 = {};
  float mrow[8], lrow[8];
#pragma unroll
  for (int r = 0; r < 8; ++r) { mrow[r] = -1e30f; lrow[r] = 0.f; }

  // preloaded K fragments for the current key block
  v16h kb00 = load_frag_b(k, DHEAD, 0,  0,  lane);
  v16h kb01 = load_frag_b(k, DHEAD, 0,  32, lane);
  v16h kb10 = load_frag_b(k, DHEAD, 16, 0,  lane);
  v16h kb11 = load_frag_b(k, DHEAD, 16, 32, lane);

  for (int jb = 0; jb < mBase + 16; jb += 32) {
    // S = Q * K^T for 32 keys (two 16-col tiles)
    v8f s0 = {}, s1 = {};
    s0 = wmma16(qa0, kb00, s0);
    s0 = wmma16(qa1, kb01, s0);
    s1 = wmma16(qa0, kb10, s1);
    s1 = wmma16(qa1, kb11, s1);

    // issue V loads for this block and K loads for the next block now,
    // so they overlap the softmax VALU section below
    v16h vb0 = load_frag_b(vt, NTOK, 0,  jb, lane);
    v16h vb1 = load_frag_b(vt, NTOK, 16, jb, lane);
    v16h vb2 = load_frag_b(vt, NTOK, 32, jb, lane);
    v16h vb3 = load_frag_b(vt, NTOK, 48, jb, lane);
    int jn = jb + 32;
    if (jn < mBase + 16) {
      kb00 = load_frag_b(k, DHEAD, jn,      0,  lane);
      kb01 = load_frag_b(k, DHEAD, jn,      32, lane);
      kb10 = load_frag_b(k, DHEAD, jn + 16, 0,  lane);
      kb11 = load_frag_b(k, DHEAD, jn + 16, 32, lane);
    }

    int c0 = jb + nloc, c1 = c0 + 16;
    bool ok0 = mk[c0] != 0, ok1 = mk[c1] != 0;

#pragma unroll
    for (int r = 0; r < 8; ++r) {
      int i = mBase + r + 8 * half;
      float a  = s0[r] * SCALEF;
      float bb = s1[r] * SCALEF;
      if (c0 > i || !ok0) a  = -1e30f;           // causal + key padding
      if (c1 > i || !ok1) bb = -1e30f;
      float mloc = fmaxf(a, bb);                 // row-max over 16-lane group
      mloc = fmaxf(mloc, __shfl_xor(mloc, 1, 32));
      mloc = fmaxf(mloc, __shfl_xor(mloc, 2, 32));
      mloc = fmaxf(mloc, __shfl_xor(mloc, 4, 32));
      mloc = fmaxf(mloc, __shfl_xor(mloc, 8, 32));
      float mnew = fmaxf(mrow[r], mloc);
      float corr = __expf(mrow[r] - mnew);
      mrow[r] = mnew;
      float p0 = __expf(a - mnew);
      float p1 = __expf(bb - mnew);
      float rs = p0 + p1;
      rs += __shfl_xor(rs, 1, 32);
      rs += __shfl_xor(rs, 2, 32);
      rs += __shfl_xor(rs, 4, 32);
      rs += __shfl_xor(rs, 8, 32);
      lrow[r] = lrow[r] * corr + rs;
      acc0[r] *= corr; acc1[r] *= corr; acc2[r] *= corr; acc3[r] *= corr;
      int row = r + 8 * half;                    // C-layout -> LDS (row-major P)
      pw[row * 32 + nloc]      = (_Float16)p0;
      pw[row * 32 + nloc + 16] = (_Float16)p1;
    }
    // P (16x32) as A-fragment straight from LDS; V already in registers
    v16h pa = load_frag_a(pw, 32, 0, 0, lane);
    acc0 = wmma16(pa, vb0, acc0);
    acc1 = wmma16(pa, vb1, acc1);
    acc2 = wmma16(pa, vb2, acc2);
    acc3 = wmma16(pa, vb3, acc3);
  }

#pragma unroll
  for (int r = 0; r < 8; ++r) {
    float inv = 1.0f / fmaxf(lrow[r], 1e-20f);
    int tok = b * NTOK + mBase + r + 8 * half;
    _Float16* orow = O + (size_t)tok * (HEADS * DHEAD) + h * DHEAD + nloc;
    orow[0]  = (_Float16)(acc0[r] * inv);
    orow[16] = (_Float16)(acc1[r] * inv);
    orow[32] = (_Float16)(acc2[r] * inv);
    orow[48] = (_Float16)(acc3[r] * inv);
  }
}

extern "C" void kernel_launch(void* const* d_in, const int* in_sizes, int n_in,
                              void* d_out, int out_size, void* d_ws, size_t ws_size,
                              hipStream_t stream) {
  const float* x            = (const float*)d_in[0];          // [2,2048,1024]
  const unsigned char* mask = (const unsigned char*)d_in[1];  // [2,2048] bool
  const float* gamma        = (const float*)d_in[2];          // [1024]
  const float* Wq           = (const float*)d_in[3];          // [1024,1024]
  const float* Wkv          = (const float*)d_in[4];          // [1024,2048]
  const float* Wo           = (const float*)d_in[5];          // [1024,1024]
  float* out = (float*)d_out;

  _Float16* ws = (_Float16*)d_ws;
  const size_t M1 = 1024ull * 1024ull;
  _Float16* XN   = ws;              // 4M f16: rmsnormed activations [4096][1024]
  _Float16* WqT  = ws + 4  * M1;    // 1M: Wq^T  [1024][1024]
  _Float16* WkvT = ws + 5  * M1;    // 2M: Wkv^T [2048][1024]
  _Float16* WoT  = ws + 7  * M1;    // 1M: Wo^T  [1024][1024]
  _Float16* Qb   = ws + 8  * M1;    // 4M: Q  [b][h][n][d]
  _Float16* Kbf  = ws + 12 * M1;    // 4M: K  [b][h][n][d]
  _Float16* Vtb  = ws + 16 * M1;    // 4M: Vt [b][h][d][n]
  _Float16* Oat  = ws + 20 * M1;    // 4M: attn out [4096][1024]

  rmsnorm_kernel<<<MROWS, 256, 0, stream>>>(x, gamma, XN);
  transpose_kernel<<<(1024 * 1024) / 256, 256, 0, stream>>>(Wq,  WqT,  1024, 1024);
  transpose_kernel<<<(1024 * 2048) / 256, 256, 0, stream>>>(Wkv, WkvT, 1024, 2048);
  transpose_kernel<<<(1024 * 1024) / 256, 256, 0, stream>>>(Wo,  WoT,  1024, 1024);
  // Q projection: (4096/256) m-blocks * (1024/64) n-blocks = 256 blocks
  gemm_kernel<<<256, 256, 0, stream>>>(XN, WqT, nullptr, Qb, nullptr, 1024, 1024, 0);
  // KV projection: 16 * 32 = 512 blocks
  gemm_kernel<<<512, 256, 0, stream>>>(XN, WkvT, nullptr, Kbf, Vtb, 1024, 2048, 1);
  // attention: 2*16*128 = 4096 waves = 512 blocks
  attn_kernel<<<512, 256, 0, stream>>>(Qb, Kbf, Vtb, mask, Oat);
  // output projection, f32 store
  gemm_kernel<<<256, 256, 0, stream>>>(Oat, WoT, out, nullptr, nullptr, 1024, 1024, 2);
}
